// GNN_82386062672575
// MI455X (gfx1250) — compile-verified, matrix operand order checked by reading, not run on hardware
//
#include <hip/hip_runtime.h>
#include <hip/hip_bf16.h>
#include <cmath>

// ---------------- sizes ----------------
#define NN    16384
#define EE    262144
#define NGRP  64
#define ETOT  (EE + NN)   // GAT adds N self-loops

typedef __attribute__((ext_vector_type(16))) __bf16 v16bf;
typedef __attribute__((ext_vector_type(8)))  __bf16 v8bf;
typedef __attribute__((ext_vector_type(8)))  float  v8f;

// ---------- order-preserving float<->uint for atomicMax ----------
__device__ __forceinline__ unsigned fkey(float f) {
  unsigned b = __float_as_uint(f);
  return (b & 0x80000000u) ? ~b : (b | 0x80000000u);
}
__device__ __forceinline__ float fdec(unsigned k) {
  unsigned b = (k & 0x80000000u) ? (k ^ 0x80000000u) : ~k;
  return __uint_as_float(b);
}
__device__ __forceinline__ float lrelu(float v) { return v > 0.f ? v : 0.2f * v; }

// ============ bf16 fragment loader: two contiguous b128 loads ============
// 16-bit A 16x32 ISA layout per lane = elements [k0+half*8 .. +7] and
// [k0+half*8+16 .. +23] of the row -> two 16-byte loads.
__device__ __forceinline__ v16bf load_frag(const __bf16* __restrict__ base, int K,
                                           int row, int k0, int half) {
  const __bf16* p = base + (size_t)row * K + k0 + (half << 3);
  v8bf lo = *(const v8bf*)p;
  v8bf hi = *(const v8bf*)(p + 16);
  return __builtin_shufflevector(lo, hi, 0, 1, 2, 3, 4, 5, 6, 7,
                                 8, 9, 10, 11, 12, 13, 14, 15);
}

__device__ __forceinline__ void store_tile(v8f c, float* __restrict__ out,
                                           __bf16* __restrict__ out_bf,
                                           const float* __restrict__ bias, int act,
                                           int Nc, int mbase, int nbase, int half,
                                           int lr) {
  const int n = nbase + lr;
  const float bv = bias ? bias[n] : 0.f;
#pragma unroll
  for (int r = 0; r < 8; ++r) {
    const int m = mbase + half * 8 + r;  // C/D layout: VGPR r -> M=r (+8 hi lanes)
    float v = c[r] + bv;
    if (act == 1) v = lrelu(v);
    else if (act == 2) v = 1.f / (1.f + __expf(-v));
    out[(size_t)m * Nc + n] = v;
    if (out_bf) out_bf[(size_t)m * Nc + n] = (__bf16)v;
  }
}

// ===== WMMA GEMM: out[M,Nc] = A[M,K](bf16) @ Wt[Nc,K](bf16,pre-transposed) =====
// 8 waves/block as 2(M) x 4(N); each wave owns a 32x32 tile (4 accumulators,
// 2 A-frags x 2 B-frags -> 4 v_wmma per K-step). Block tile = 64 x 128.
__global__ __launch_bounds__(256) void gemm_wmma(
    const __bf16* __restrict__ A, const __bf16* __restrict__ Wt,
    const float* __restrict__ bias, float* __restrict__ out,
    __bf16* __restrict__ out_bf, int M, int K, int Nc, int act) {
  const int lane = threadIdx.x & 31;
  const int wave = threadIdx.x >> 5;
  const int wm = wave >> 2, wn = wave & 3;
  const int m0 = blockIdx.y * 64 + wm * 32;
  const int n0 = blockIdx.x * 128 + wn * 32;
  const int half = lane >> 4, lr = lane & 15;

  v8f c00 = {0.f, 0.f, 0.f, 0.f, 0.f, 0.f, 0.f, 0.f};
  v8f c01 = {0.f, 0.f, 0.f, 0.f, 0.f, 0.f, 0.f, 0.f};
  v8f c10 = {0.f, 0.f, 0.f, 0.f, 0.f, 0.f, 0.f, 0.f};
  v8f c11 = {0.f, 0.f, 0.f, 0.f, 0.f, 0.f, 0.f, 0.f};

  for (int k0 = 0; k0 < K; k0 += 32) {
    __builtin_prefetch(A + (size_t)(m0 + lr) * K + k0 + 64, 0, 0);
    v16bf a0 = load_frag(A, K, m0 + lr, k0, half);
    v16bf a1 = load_frag(A, K, m0 + 16 + lr, k0, half);
    v16bf b0 = load_frag(Wt, K, n0 + lr, k0, half);
    v16bf b1 = load_frag(Wt, K, n0 + 16 + lr, k0, half);
    c00 = __builtin_amdgcn_wmma_f32_16x16x32_bf16(false, a0, false, b0, (short)0, c00, false, false);
    c01 = __builtin_amdgcn_wmma_f32_16x16x32_bf16(false, a0, false, b1, (short)0, c01, false, false);
    c10 = __builtin_amdgcn_wmma_f32_16x16x32_bf16(false, a1, false, b0, (short)0, c10, false, false);
    c11 = __builtin_amdgcn_wmma_f32_16x16x32_bf16(false, a1, false, b1, (short)0, c11, false, false);
  }

  store_tile(c00, out, out_bf, bias, act, Nc, m0,      n0,      half, lr);
  store_tile(c01, out, out_bf, bias, act, Nc, m0,      n0 + 16, half, lr);
  store_tile(c10, out, out_bf, bias, act, Nc, m0 + 16, n0,      half, lr);
  store_tile(c11, out, out_bf, bias, act, Nc, m0 + 16, n0 + 16, half, lr);
}

// ================= conversion kernels =================
__global__ void cvt_bf16_kernel(const float* __restrict__ in, __bf16* out, int n) {
  int i = blockIdx.x * blockDim.x + threadIdx.x;
  if (i < n) out[i] = (__bf16)in[i];
}

// W[K,Nc] f32 -> Wt[Nc,K] bf16 (transpose + convert, done once per GEMM)
__global__ void wt_cvt_kernel(const float* __restrict__ W, __bf16* out, int K, int Nc) {
  int i = blockIdx.x * blockDim.x + threadIdx.x;
  if (i >= K * Nc) return;
  int k = i / Nc, n = i % Nc;
  out[(size_t)n * K + k] = (__bf16)W[i];
}

// ================= element / fill kernels =================
__global__ void fill_u32_kernel(unsigned* p, unsigned v, int n) {
  int i = blockIdx.x * blockDim.x + threadIdx.x;
  if (i < n) p[i] = v;
}

__global__ void add_lrelu_kernel(const float* a, const float* b, float* out,
                                 __bf16* out_bf, int n) {
  int i = blockIdx.x * blockDim.x + threadIdx.x;
  if (i < n) {
    float v = lrelu(a[i] + b[i]);
    out[i] = v;
    out_bf[i] = (__bf16)v;
  }
}

// ================= GAT attention pieces =================
__global__ void gat_node_sums(const float* __restrict__ pre,
                              const float* __restrict__ as,
                              const float* __restrict__ ad,
                              float* ssum, float* dsum, int n_nodes, int H, int C) {
  int idx = blockIdx.x * blockDim.x + threadIdx.x;
  if (idx >= n_nodes * H) return;
  int n = idx / H, h = idx % H;
  const float* row = pre + (size_t)n * H * C + h * C;
  const float* a1 = as + h * C;
  const float* a2 = ad + h * C;
  float s = 0.f, d = 0.f;
  for (int c = 0; c < C; ++c) { float v = row[c]; s += v * a1[c]; d += v * a2[c]; }
  ssum[idx] = s; dsum[idx] = d;
}

__global__ void gat_edge_logits(const int* __restrict__ src, const int* __restrict__ dst,
                                int nE, int nTot, const float* __restrict__ ssum,
                                const float* __restrict__ dsum, float* eb,
                                unsigned* maxk, int H) {
  int i = blockIdx.x * blockDim.x + threadIdx.x;
  if (i >= nTot) return;
  int s = (i < nE) ? src[i] : (i - nE);
  int d = (i < nE) ? dst[i] : (i - nE);
  for (int h = 0; h < H; ++h) {
    float e = lrelu(ssum[s * H + h] + dsum[d * H + h]);
    eb[(size_t)i * H + h] = e;
    atomicMax(&maxk[d * H + h], fkey(e));
  }
}

__global__ void tconv_edge_logits(const int* __restrict__ src, const int* __restrict__ dst,
                                  int nE, const float* __restrict__ q,
                                  const float* __restrict__ k, float rscale,
                                  float* eb, unsigned* maxk, int H, int C) {
  int i = blockIdx.x * blockDim.x + threadIdx.x;
  if (i >= nE) return;
  int s = src[i], d = dst[i];
  for (int h = 0; h < H; ++h) {
    const float* qr = q + (size_t)d * H * C + h * C;
    const float* kr = k + (size_t)s * H * C + h * C;
    float acc = 0.f;
    for (int c = 0; c < C; ++c) acc += qr[c] * kr[c];
    acc *= rscale;
    eb[(size_t)i * H + h] = acc;
    atomicMax(&maxk[d * H + h], fkey(acc));
  }
}

__global__ void edge_expsum(const int* __restrict__ dst, int nE, int nTot,
                            float* eb, const unsigned* __restrict__ maxk,
                            float* sumb, int H) {
  int i = blockIdx.x * blockDim.x + threadIdx.x;
  if (i >= nTot) return;
  int d = (i < nE) ? dst[i] : (i - nE);
  for (int h = 0; h < H; ++h) {
    float ex = __expf(eb[(size_t)i * H + h] - fdec(maxk[d * H + h]));
    eb[(size_t)i * H + h] = ex;
    atomicAdd(&sumb[d * H + h], ex);
  }
}

__global__ void edge_aggregate(const int* __restrict__ src, const int* __restrict__ dst,
                               int nE, const float* __restrict__ val,
                               const float* __restrict__ eb,
                               const float* __restrict__ sumb,
                               float* out, int H, int C) {
  int i = blockIdx.x;
  int s = (i < nE) ? src[i] : (i - nE);
  int d = (i < nE) ? dst[i] : (i - nE);
  int HC = H * C;
  for (int ch = threadIdx.x; ch < HC; ch += blockDim.x) {
    int h = ch / C;
    float al = eb[(size_t)i * H + h] / (sumb[(size_t)d * H + h] + 1e-16f);
    atomicAdd(&out[(size_t)d * HC + ch], val[(size_t)s * HC + ch] * al);
  }
}

// ================= BatchNorm =================
__global__ void bn_stats(const float* __restrict__ x, int n_nodes, int F,
                         float* mean, float* rstd) {
  __shared__ float s1[256], s2[256];
  int j = blockIdx.x;
  float a = 0.f, b = 0.f;
  for (int n = threadIdx.x; n < n_nodes; n += blockDim.x) {
    float v = x[(size_t)n * F + j];
    a += v; b += v * v;
  }
  s1[threadIdx.x] = a; s2[threadIdx.x] = b;
  __syncthreads();
  for (int t = 128; t > 0; t >>= 1) {
    if ((int)threadIdx.x < t) { s1[threadIdx.x] += s1[threadIdx.x + t];
                                s2[threadIdx.x] += s2[threadIdx.x + t]; }
    __syncthreads();
  }
  if (threadIdx.x == 0) {
    float m = s1[0] / n_nodes;
    float v = s2[0] / n_nodes - m * m;
    mean[j] = m; rstd[j] = rsqrtf(v + 1e-5f);
  }
}

__global__ void bn_apply_lrelu(const float* __restrict__ x, const float* mean,
                               const float* rstd, const float* g, const float* be,
                               float* out, __bf16* out_bf, int tot, int F) {
  int i = blockIdx.x * blockDim.x + threadIdx.x;
  if (i >= tot) return;
  int j = i % F;
  float v = lrelu(g[j] * (x[i] - mean[j]) * rstd[j] + be[j]);
  out[i] = v;
  out_bf[i] = (__bf16)v;
}

// mean over heads: out[n,c] = (1/H) sum_h in[n, h*C + c]
__global__ void head_mean(const float* __restrict__ in, float* out, int n_nodes,
                          int H, int C) {
  int i = blockIdx.x * blockDim.x + threadIdx.x;
  if (i >= n_nodes * C) return;
  int n = i / C, c = i % C;
  float acc = 0.f;
  for (int h = 0; h < H; ++h) acc += in[(size_t)n * H * C + h * C + c];
  out[i] = acc / H;
}

// h5 = lrelu(mean_heads(agg[N,4*256]) + skip) + ident
__global__ void tc2_final(const float* __restrict__ agg, const float* __restrict__ skip,
                          const float* __restrict__ ident, float* out, int n_nodes) {
  int i = blockIdx.x * blockDim.x + threadIdx.x;
  if (i >= n_nodes * 256) return;
  int n = i / 256, c = i % 256;
  const float* r = agg + (size_t)n * 1024 + c;
  float m = (r[0] + r[256] + r[512] + r[768]) * 0.25f;
  out[i] = lrelu(m + skip[i]) + ident[i];
}

// ================= pooling + layernorm =================
__global__ void pool_atomic(const float* __restrict__ h, const int* __restrict__ batch,
                            unsigned* gmaxk, float* gsum, float* cnt, int n_nodes) {
  int i = blockIdx.x * blockDim.x + threadIdx.x;
  if (i >= n_nodes * 256) return;
  int n = i / 256, c = i % 256;
  int g = batch[n];
  float v = h[i];
  atomicMax(&gmaxk[g * 256 + c], fkey(v));
  atomicAdd(&gsum[g * 256 + c], v);
  if (c == 0) atomicAdd(&cnt[g], 1.0f);
}

__global__ __launch_bounds__(512) void pool_ln(const unsigned* __restrict__ gmaxk,
                                               const float* __restrict__ gsum,
                                               const float* __restrict__ cnt,
                                               const float* lng, const float* lnb,
                                               float* z, __bf16* zbf) {
  __shared__ float red[512];
  int g = blockIdx.x, t = threadIdx.x;
  float v = (t < 256) ? fdec(gmaxk[g * 256 + t]) : (gsum[g * 256 + (t - 256)] / cnt[g]);
  red[t] = v; __syncthreads();
  for (int s = 256; s > 0; s >>= 1) { if (t < s) red[t] += red[t + s]; __syncthreads(); }
  float mu = red[0] * (1.f / 512.f);
  __syncthreads();
  red[t] = (v - mu) * (v - mu); __syncthreads();
  for (int s = 256; s > 0; s >>= 1) { if (t < s) red[t] += red[t + s]; __syncthreads(); }
  float var = red[0] * (1.f / 512.f);
  float y = lng[t] * (v - mu) * rsqrtf(var + 1e-5f) + lnb[t];
  z[(size_t)g * 512 + t] = y;
  zbf[(size_t)g * 512 + t] = (__bf16)y;
}

// ======================= host driver =======================
extern "C" void kernel_launch(void* const* d_in, const int* in_sizes, int n_in,
                              void* d_out, int out_size, void* d_ws, size_t ws_size,
                              hipStream_t stream) {
  (void)in_sizes; (void)n_in; (void)out_size; (void)ws_size;
  const float* x    = (const float*)d_in[0];
  const int*   ei   = (const int*)d_in[1];
  const int*   srcp = ei;            // edge_index[0]
  const int*   dstp = ei + EE;       // edge_index[1]
  const int*   batch= (const int*)d_in[2];
  const float* W1 = (const float*)d_in[3];
  const float* as1= (const float*)d_in[4];
  const float* ad1= (const float*)d_in[5];
  const float* g1 = (const float*)d_in[7];
  const float* be1= (const float*)d_in[8];
  const float* W2 = (const float*)d_in[9];
  const float* as2= (const float*)d_in[10];
  const float* ad2= (const float*)d_in[11];
  const float* g2 = (const float*)d_in[13];
  const float* be2= (const float*)d_in[14];
  const float* W3 = (const float*)d_in[15];
  const float* as3= (const float*)d_in[16];
  const float* ad3= (const float*)d_in[17];
  const float* g3 = (const float*)d_in[19];
  const float* be3= (const float*)d_in[20];
  const float* Wq1= (const float*)d_in[21]; const float* bq1=(const float*)d_in[22];
  const float* Wk1= (const float*)d_in[23]; const float* bk1=(const float*)d_in[24];
  const float* Wv1= (const float*)d_in[25]; const float* bv1=(const float*)d_in[26];
  const float* Ws1= (const float*)d_in[27]; const float* bs1=(const float*)d_in[28];
  const float* Wq2= (const float*)d_in[29]; const float* bq2=(const float*)d_in[30];
  const float* Wk2= (const float*)d_in[31]; const float* bk2=(const float*)d_in[32];
  const float* Wv2= (const float*)d_in[33]; const float* bv2=(const float*)d_in[34];
  const float* Ws2= (const float*)d_in[35]; const float* bs2=(const float*)d_in[36];
  const float* lng= (const float*)d_in[37]; const float* lnb=(const float*)d_in[38];
  const float* fW1= (const float*)d_in[39]; const float* fb1=(const float*)d_in[40];
  const float* fW2= (const float*)d_in[41]; const float* fb2=(const float*)d_in[42];
  float* outp = (float*)d_out;

  // ---- workspace layout (float-unit allocator; bf16 arenas are casts) ----
  float* w = (float*)d_ws;
  size_t off = 0;
  auto alloc = [&](size_t n) { float* p = w + off; off += n; return p; };
  float* P0 = alloc((size_t)NN * 1024);  // wide GEMM outputs (pre / q)
  float* P1 = alloc((size_t)NN * 1024);  // out1 / k
  float* P2 = alloc((size_t)NN * 1024);  // out2 / v
  float* P3 = alloc((size_t)NN * 1024);  // attention aggregation target
  float* P4 = alloc((size_t)NN * 512);   // skip buffers
  float* P5 = alloc((size_t)NN * 512);   // h4
  float* P6 = alloc((size_t)NN * 256);   // h3 (ident)
  float* P7 = alloc((size_t)NN * 256);   // h5
  float* EB = alloc((size_t)ETOT * 8);   // per-edge logits / exp
  float* SS = alloc((size_t)NN * 8);     // GAT ssum
  float* DD = alloc((size_t)NN * 8);     // GAT dsum
  float* MXf= alloc((size_t)NN * 8);     // segment max keys (as u32)
  float* SM = alloc((size_t)NN * 8);     // segment sums
  float* MEAN = alloc(1024);
  float* RSTD = alloc(1024);
  float* Z  = alloc((size_t)NGRP * 512);
  float* Z2 = alloc((size_t)NGRP * 512);
  float* GMAXf = alloc((size_t)NGRP * 256);
  float* GSUM  = alloc((size_t)NGRP * 256);
  float* CNT   = alloc(NGRP);
  __bf16* ABF  = (__bf16*)alloc((size_t)NN * 256);    // N x 512 bf16 activation arena
  __bf16* WBF  = (__bf16*)alloc((size_t)512 * 512);   // 512x1024 bf16 weight arena
  __bf16* ZBF  = (__bf16*)alloc((size_t)NGRP * 256);  // 64x512 bf16
  __bf16* ZBF2 = (__bf16*)alloc((size_t)NGRP * 256);  // 64x512 bf16
  unsigned* MX   = (unsigned*)MXf;
  unsigned* GMAX = (unsigned*)GMAXf;

  auto fill0 = [&](void* p, size_t n) {
    fill_u32_kernel<<<(int)((n + 255) / 256), 256, 0, stream>>>((unsigned*)p, 0u, (int)n);
  };
  // transpose-convert weight then run WMMA GEMM
  auto gemm = [&](const __bf16* A, const float* Wf, const float* bias, float* out,
                  __bf16* out_bf, int M, int K, int Nc, int act) {
    wt_cvt_kernel<<<(K * Nc + 255) / 256, 256, 0, stream>>>(Wf, WBF, K, Nc);
    dim3 grid(Nc / 128, M / 64);
    gemm_wmma<<<grid, 256, 0, stream>>>(A, WBF, bias, out, out_bf, M, K, Nc, act);
  };
  auto bn_layer = [&](const float* xin, const float* g, const float* be, float* out,
                      __bf16* out_bf, int F) {
    bn_stats<<<F, 256, 0, stream>>>(xin, NN, F, MEAN, RSTD);
    int tot = NN * F;
    bn_apply_lrelu<<<(tot + 255) / 256, 256, 0, stream>>>(xin, MEAN, RSTD, g, be, out,
                                                          out_bf, tot, F);
  };
  auto gat_att = [&](const float* pre, const float* as, const float* ad, int H, int C,
                     float* agg) {
    gat_node_sums<<<(NN * H + 255) / 256, 256, 0, stream>>>(pre, as, ad, SS, DD, NN, H, C);
    fill0(MX, (size_t)NN * H);
    fill0(SM, (size_t)NN * H);
    fill0(agg, (size_t)NN * H * C);
    gat_edge_logits<<<(ETOT + 255) / 256, 256, 0, stream>>>(srcp, dstp, EE, ETOT, SS, DD,
                                                            EB, MX, H);
    edge_expsum<<<(ETOT + 255) / 256, 256, 0, stream>>>(dstp, EE, ETOT, EB, MX, SM, H);
    edge_aggregate<<<ETOT, 256, 0, stream>>>(srcp, dstp, EE, pre, EB, SM, agg, H, C);
  };
  auto tconv_att = [&](const float* q, const float* k, const float* v, int H, int C,
                       float* agg) {
    fill0(MX, (size_t)NN * H);
    fill0(SM, (size_t)NN * H);
    fill0(agg, (size_t)NN * H * C);
    float rscale = 1.0f / sqrtf((float)C);
    tconv_edge_logits<<<(EE + 255) / 256, 256, 0, stream>>>(srcp, dstp, EE, q, k, rscale,
                                                            EB, MX, H, C);
    edge_expsum<<<(EE + 255) / 256, 256, 0, stream>>>(dstp, EE, EE, EB, MX, SM, H);
    edge_aggregate<<<EE, 256, 0, stream>>>(srcp, dstp, EE, v, EB, SM, agg, H, C);
  };

  // ---- GAT1: 128 -> 256 (H=8,C=32)  [bias b1 cancels under BN] ----
  cvt_bf16_kernel<<<(NN * 128 + 255) / 256, 256, 0, stream>>>(x, ABF, NN * 128);
  gemm(ABF, W1, nullptr, P0, nullptr, NN, 128, 256, 0);
  gat_att(P0, as1, ad1, 8, 32, P3);
  bn_layer(P3, g1, be1, P1, ABF, 256);            // out1 = P1 [N,256] (+bf16)

  // ---- GAT2: 256 -> 512 (H=8,C=64) ----
  gemm(ABF, W2, nullptr, P0, nullptr, NN, 256, 512, 0);
  gat_att(P0, as2, ad2, 8, 64, P3);
  bn_layer(P3, g2, be2, P2, ABF, 512);            // out2 = P2 [N,512] (+bf16)

  // ---- GAT3: 512 -> 1024, mean over 4 heads -> 256 ----
  gemm(ABF, W3, nullptr, P0, nullptr, NN, 512, 1024, 0);
  gat_att(P0, as3, ad3, 4, 256, P3);
  head_mean<<<(NN * 256 + 255) / 256, 256, 0, stream>>>(P3, P6, NN, 4, 256);
  bn_layer(P6, g3, be3, P6, ABF, 256);            // h3 = ident = P6 (+bf16)

  // ---- TransformerConv1: 256 -> 512 (H=8,C=64, concat) ----
  gemm(ABF, Wq1, bq1, P0, nullptr, NN, 256, 512, 0);   // q1
  gemm(ABF, Wk1, bk1, P1, nullptr, NN, 256, 512, 0);   // k1
  gemm(ABF, Wv1, bv1, P2, nullptr, NN, 256, 512, 0);   // v1
  gemm(ABF, Ws1, bs1, P4, nullptr, NN, 256, 512, 0);   // skip1
  tconv_att(P0, P1, P2, 8, 64, P3);
  add_lrelu_kernel<<<(NN * 512 + 255) / 256, 256, 0, stream>>>(P3, P4, P5, ABF,
                                                               NN * 512);  // h4 (+bf16)

  // ---- TransformerConv2: 512 -> 1024, mean 4 heads -> 256, skip 512->256 ----
  gemm(ABF, Wq2, bq2, P0, nullptr, NN, 512, 1024, 0);  // q2
  gemm(ABF, Wk2, bk2, P1, nullptr, NN, 512, 1024, 0);  // k2
  gemm(ABF, Wv2, bv2, P2, nullptr, NN, 512, 1024, 0);  // v2
  gemm(ABF, Ws2, bs2, P4, nullptr, NN, 512, 256, 0);   // skip2
  tconv_att(P0, P1, P2, 4, 256, P3);
  tc2_final<<<(NN * 256 + 255) / 256, 256, 0, stream>>>(P3, P4, P6, P7, NN);  // h5

  // ---- pooling + layernorm ----
  fill0(GMAX, (size_t)NGRP * 256);
  fill0(GSUM, (size_t)NGRP * 256);
  fill0(CNT, NGRP);
  pool_atomic<<<(NN * 256 + 255) / 256, 256, 0, stream>>>(P7, batch, GMAX, GSUM, CNT, NN);
  pool_ln<<<NGRP, 512, 0, stream>>>(GMAX, GSUM, CNT, lng, lnb, Z, ZBF);

  // ---- MLP head (WMMA): 64x512 @ 512x512 lrelu ; 64x512 @ 512x256 sigmoid ----
  gemm(ZBF, fW1, fb1, Z2, ZBF2, NGRP, 512, 512, 1);
  gemm(ZBF2, fW2, fb2, outp, nullptr, NGRP, 512, 256, 2);
}
